// MultiheadAttention_67534065762563
// MI455X (gfx1250) — matrix operand order repaired and checked by reading, not compiled, not run on hardware
//
#include <hip/hip_runtime.h>

// MI455X / gfx1250 multi-head attention.
// GEMMs on v_wmma_f32_16x16x32_f16 (f16 in, f32 accumulate), flash-style online
// softmax, K/V tiles streamed into LDS by the Tensor Data Mover (TENSORcnt).

typedef __attribute__((ext_vector_type(16))) _Float16 v16h;
typedef __attribute__((ext_vector_type(8)))  _Float16 v8h;
typedef __attribute__((ext_vector_type(8)))  float    v8f;
typedef __attribute__((ext_vector_type(4))) unsigned int u32x4;
typedef __attribute__((ext_vector_type(4))) int          i32x4;
typedef __attribute__((ext_vector_type(8))) int          i32x8;

#define S_LEN 2048
#define DM    1024
#define NH    16
#define HD    64

__device__ __forceinline__ v8f wmma_f16(v16h a, v16h b, v8f c) {
  return __builtin_amdgcn_wmma_f32_16x16x32_f16(
      /*neg_a=*/false, a, /*neg_b=*/false, b,
      /*c_mod=*/(short)0, c, /*reuse_a=*/false, /*reuse_b=*/false);
}

// A-fragment (16x32 f16): lane holds row = lane&15, K = {kb..kb+7, 16+kb..16+kb+7},
// kb = (lane>=16)?8:0. Source row-major, leading dim ld (elements).
__device__ __forceinline__ v16h ld_fragA(const _Float16* p, int ld) {
  int lane = threadIdx.x & 31;
  int r  = lane & 15;
  int kb = (lane & 16) ? 8 : 0;
  const _Float16* q = p + r * ld + kb;
  v8h lo = *(const v8h*)(q);
  v8h hi = *(const v8h*)(q + 16);
  v16h f;
#pragma unroll
  for (int i = 0; i < 8; ++i) { f[i] = lo[i]; f[i + 8] = hi[i]; }
  return f;
}

// B-fragment (32x16 f16): lane holds col = lane&15, K = {kb..kb+7, 16+kb..16+kb+7},
// kb = (lane>=16)?8:0. Source stored "column-major as rows": p[n*ld + k].
__device__ __forceinline__ v16h ld_fragB(const _Float16* p, int ld) {
  int lane = threadIdx.x & 31;
  int n  = lane & 15;
  int kb = (lane & 16) ? 8 : 0;
  const _Float16* q = p + n * ld + kb;
  v8h lo = *(const v8h*)(q);
  v8h hi = *(const v8h*)(q + 16);
  v16h f;
#pragma unroll
  for (int i = 0; i < 8; ++i) { f[i] = lo[i]; f[i + 8] = hi[i]; }
  return f;
}

// ---- Tensor Data Mover: 1D/2D tile (f16 elements) from global into LDS ----
// Builds a D# per CDNA5 ISA: group0 = {count=1, lds_addr,
// global_addr, type=2}; group1 = {data_size=2B, huge tensor dims (no OOB clip),
// tile_dim0/1, 48-bit dim0 stride}. tile_d1 == 0 -> 1D contiguous copy.
__device__ __forceinline__ void tdm_load_tile(unsigned int ldsoff, const void* gptr,
                                              unsigned int tile_d0, unsigned int tile_d1,
                                              unsigned long long stride0_elems) {
  unsigned long long ga = (unsigned long long)(uintptr_t)gptr;
  u32x4 g0;
  g0[0] = 1u;                                        // count=1, is_restore=0
  g0[1] = ldsoff;                                    // lds_addr (bytes)
  g0[2] = (unsigned int)ga;                          // global_addr[31:0]
  g0[3] = (unsigned int)((ga >> 32) & 0x01FFFFFFu)   // global_addr[56:32]
          | (2u << 30);                              // type = 2 ("image")
  i32x8 g1;
  const unsigned int td = 0x7FFFFFFFu;               // tensor dims: effectively inf
  g1[0] = 1 << 16;                                   // data_size=1 -> 2 bytes/elem
  g1[1] = (int)((td & 0xFFFFu) << 16);                               // dim0[15:0]
  g1[2] = (int)(((td >> 16) & 0xFFFFu) | ((td & 0xFFFFu) << 16));    // dim0[31:16],dim1[15:0]
  g1[3] = (int)(((td >> 16) & 0xFFFFu) | ((tile_d0 & 0xFFFFu) << 16)); // dim1[31:16],tile0
  g1[4] = (int)(tile_d1 & 0xFFFFu);                                  // tile1 (tile2=0)
  g1[5] = (int)(unsigned int)(stride0_elems & 0xFFFFFFFFu);          // stride0[31:0]
  g1[6] = (int)(unsigned int)((stride0_elems >> 32) & 0xFFFFu);      // stride0[47:32]
  g1[7] = 0;
  i32x4 z4 = {0, 0, 0, 0};
#if __clang_major__ >= 23
  i32x8 z8 = {0, 0, 0, 0, 0, 0, 0, 0};
  __builtin_amdgcn_tensor_load_to_lds(g0, g1, z4, z4, z8, 0);
#else
  __builtin_amdgcn_tensor_load_to_lds(g0, g1, z4, z4, 0);
#endif
}

__device__ __forceinline__ unsigned int lds_off(const void* p) {
  // LDS aperture: flat addr low 32 bits == LDS byte offset (ISA §10.2).
  return (unsigned int)(uintptr_t)p;
}

// ---------------- prep kernels ----------------
__global__ void cvt_f16(const float* __restrict__ src, _Float16* __restrict__ dst, int n) {
  int i = blockIdx.x * blockDim.x + threadIdx.x;
  if (i < n) dst[i] = (_Float16)src[i];
}

// dst[c][r] = (f16) src[r][c]   (src: rows x cols)
__global__ void transpose_cvt(const float* __restrict__ src, _Float16* __restrict__ dst,
                              int rows, int cols) {
  int i = blockIdx.x * blockDim.x + threadIdx.x;
  if (i < rows * cols) {
    int r = i / cols, c = i % cols;
    dst[c * rows + r] = (_Float16)src[i];
  }
}

// ---------------- QKV projection: [4096,1024] x [1024,3072] + bias ----------------
// Writes Q,K as [b*NH + h][s][HD] f16 and V transposed as [b*NH + h][HD][s] f16.
__global__ void __launch_bounds__(128) qkv_gemm(const _Float16* __restrict__ X,
                                                const _Float16* __restrict__ WT,  // [3072][1024]
                                                const float* __restrict__ bqkv,
                                                _Float16* __restrict__ Q,
                                                _Float16* __restrict__ K,
                                                _Float16* __restrict__ Vt) {
  int wid  = blockIdx.x * 4 + (threadIdx.x >> 5);
  int lane = threadIdx.x & 31;
  int mt = wid / 48;           // 256 row tiles
  int ng = wid % 48;           // 48 groups of 64 cols
  int m0 = mt * 16;
  int n0 = ng * 64;

  v8f acc[4] = {v8f{}, v8f{}, v8f{}, v8f{}};
  for (int k0 = 0; k0 < DM; k0 += 32) {
    v16h a = ld_fragA(X + (size_t)m0 * DM + k0, DM);
#pragma unroll
    for (int j = 0; j < 4; ++j) {
      v16h b = ld_fragB(WT + (size_t)(n0 + 16 * j) * DM + k0, DM);
      acc[j] = wmma_f16(a, b, acc[j]);
    }
  }

  int hi8 = (lane & 16) ? 8 : 0;
  int nl  = lane & 15;
#pragma unroll
  for (int j = 0; j < 4; ++j) {
    int cn   = n0 + 16 * j;          // absolute output column
    int head = cn / 192;
    int t    = cn % 192;
    int seg  = t / 64;               // 0=Q 1=K 2=V
    int off  = (t % 64) + nl;        // column inside head slice
    float bias = bqkv[cn + nl];
#pragma unroll
    for (int r = 0; r < 8; ++r) {
      int m  = m0 + r + hi8;
      int bb = m >> 11;              // / S_LEN
      int s  = m & (S_LEN - 1);
      float v = acc[j][r] + bias;
      size_t bh = (size_t)(bb * NH + head);
      if (seg == 0)      Q [(bh * S_LEN + s) * HD + off] = (_Float16)v;
      else if (seg == 1) K [(bh * S_LEN + s) * HD + off] = (_Float16)v;
      else               Vt[(bh * HD + off) * S_LEN + s] = (_Float16)v;
    }
  }
}

// ---------------- flash attention ----------------
// One wave per 16-query tile per (b,h); K/V tiles double-buffered in LDS via TDM.
__global__ void __launch_bounds__(64) attn(const _Float16* __restrict__ Q,
                                           const _Float16* __restrict__ K,
                                           const _Float16* __restrict__ Vt,
                                           const float* __restrict__ mask,
                                           _Float16* __restrict__ O) {
  __shared__ __align__(16) _Float16 plds[2][16 * 32];        // P tile per wave
  __shared__ __align__(16) _Float16 kstage[2][2][32 * 64];   // [wave][buf]: 32 keys x 64 hd
  __shared__ __align__(16) _Float16 vstage[2][2][64 * 32];   // [wave][buf]: 64 hd x 32 keys

  int w    = threadIdx.x >> 5;
  int lane = threadIdx.x & 31;
  int wid  = blockIdx.x * 2 + w;
  int bh = wid >> 7;                 // 32 (b,h) pairs
  int mt = wid & 127;                // 128 query tiles
  int q0 = mt * 16;

  const _Float16* Qb  = Q  + ((size_t)bh * S_LEN + q0) * HD;
  const _Float16* Kbh = K  + (size_t)bh * S_LEN * HD;
  const _Float16* Vbh = Vt + (size_t)bh * HD * S_LEN;

  v16h a0 = ld_fragA(Qb, HD);        // hd 0..31
  v16h a1 = ld_fragA(Qb + 32, HD);   // hd 32..63

  float mrow[8], lrow[8];
  v8f o[4] = {v8f{}, v8f{}, v8f{}, v8f{}};
#pragma unroll
  for (int r = 0; r < 8; ++r) { mrow[r] = -3.0e38f; lrow[r] = 0.f; }

  int hi8 = (lane & 16) ? 8 : 0;
  int nl  = lane & 15;
  const float scale = 0.125f;        // 1/sqrt(64)

  // Prologue: async-stage tile 0 (2 TDM descriptors -> TENSORcnt ops, in-order).
  tdm_load_tile(lds_off(&kstage[w][0][0]), Kbh, 32 * HD, 0, 0);          // 1D, 4KB
  tdm_load_tile(lds_off(&vstage[w][0][0]), Vbh, 32, HD, S_LEN);          // 2D strided

  for (int it = 0; it < S_LEN / 32; ++it) {
    int K0  = it * 32;
    int cur = it & 1;
    if (K0 + 32 < S_LEN) {
      int nxt = cur ^ 1;
      tdm_load_tile(lds_off(&kstage[w][nxt][0]), Kbh + (size_t)(K0 + 32) * HD,
                    32 * HD, 0, 0);
      tdm_load_tile(lds_off(&vstage[w][nxt][0]), Vbh + (K0 + 32),
                    32, HD, S_LEN);
      __builtin_amdgcn_s_wait_tensorcnt(2);   // current tile's 2 ops complete
    } else {
      __builtin_amdgcn_s_wait_tensorcnt(0);
    }
    __builtin_amdgcn_wave_barrier();

    const _Float16* kb = &kstage[w][cur][0];  // [32][64]
    const _Float16* vb = &vstage[w][cur][0];  // [64][32]

    // scores for keys K0..K0+31 (two 16-col subtiles) from LDS
    v16h bk00 = ld_fragB(kb,                HD);
    v16h bk01 = ld_fragB(kb + 32,           HD);
    v16h bk10 = ld_fragB(kb + 16 * HD,      HD);
    v16h bk11 = ld_fragB(kb + 16 * HD + 32, HD);
    v8f s0 = v8f{}, s1 = v8f{};
    s0 = wmma_f16(a0, bk00, s0); s0 = wmma_f16(a1, bk01, s0);
    s1 = wmma_f16(a0, bk10, s1); s1 = wmma_f16(a1, bk11, s1);

    // scale + additive mask (global, L2 resident)
    const float* mp = mask + (size_t)(q0 + hi8) * S_LEN + K0 + nl;
#pragma unroll
    for (int r = 0; r < 8; ++r) {
      s0[r] = s0[r] * scale + mp[(size_t)r * S_LEN];
      s1[r] = s1[r] * scale + mp[(size_t)r * S_LEN + 16];
    }

    // online softmax; stash P into LDS as f16 (D-layout)
#pragma unroll
    for (int r = 0; r < 8; ++r) {
      float mx = fmaxf(s0[r], s1[r]);
#pragma unroll
      for (int d = 8; d >= 1; d >>= 1) mx = fmaxf(mx, __shfl_xor(mx, d, 16));
      float mn    = fmaxf(mrow[r], mx);
      float alpha = __expf(mrow[r] - mn);
      mrow[r] = mn;
      float p0 = __expf(s0[r] - mn);
      float p1 = __expf(s1[r] - mn);
      lrow[r] = lrow[r] * alpha + p0 + p1;     // per-lane partial row sum
#pragma unroll
      for (int j = 0; j < 4; ++j) o[j][r] *= alpha;
      plds[w][(r + hi8) * 32 + nl]      = (_Float16)p0;
      plds[w][(r + hi8) * 32 + 16 + nl] = (_Float16)p1;
    }
    __builtin_amdgcn_wave_barrier();           // same-wave LDS ops are ordered in HW

    // re-read P as an A-fragment (16x32)
    v16h pa;
    {
      int kb2 = (lane & 16) ? 8 : 0;
      v8h lo = *(const v8h*)&plds[w][nl * 32 + kb2];
      v8h hi = *(const v8h*)&plds[w][nl * 32 + 16 + kb2];
#pragma unroll
      for (int i = 0; i < 8; ++i) { pa[i] = lo[i]; pa[i + 8] = hi[i]; }
    }
    __builtin_amdgcn_wave_barrier();

    // O += P @ V  (V tile compacted in LDS as [64 hd][32 keys])
#pragma unroll
    for (int j = 0; j < 4; ++j) {
      v16h bv = ld_fragB(vb + (size_t)(16 * j) * 32, 32);
      o[j] = wmma_f16(pa, bv, o[j]);
    }
  }

  // finalize: reduce row sums across the 16 lanes holding each row
#pragma unroll
  for (int r = 0; r < 8; ++r) {
    float l = lrow[r];
#pragma unroll
    for (int d = 8; d >= 1; d >>= 1) l += __shfl_xor(l, d, 16);
    float inv = 1.0f / l;
#pragma unroll
    for (int j = 0; j < 4; ++j) o[j][r] *= inv;
  }

  int h  = bh & (NH - 1);
  int bb = bh >> 4;
#pragma unroll
  for (int j = 0; j < 4; ++j) {
#pragma unroll
    for (int r = 0; r < 8; ++r) {
      size_t row = (size_t)bb * S_LEN + q0 + r + hi8;
      O[row * DM + h * HD + 16 * j + nl] = (_Float16)o[j][r];
    }
  }
}

// ---------------- output projection: [4096,1024] x [1024,1024] + bias ----------------
__global__ void __launch_bounds__(128) out_gemm(const _Float16* __restrict__ Of16,
                                                const _Float16* __restrict__ WT, // [1024][1024]
                                                const float* __restrict__ bout,
                                                float* __restrict__ Y) {
  int wid  = blockIdx.x * 4 + (threadIdx.x >> 5);
  int lane = threadIdx.x & 31;
  int mt = wid >> 4;                 // 256 row tiles
  int ng = wid & 15;                 // 16 groups of 64 cols
  int m0 = mt * 16, n0 = ng * 64;

  v8f acc[4] = {v8f{}, v8f{}, v8f{}, v8f{}};
  for (int k0 = 0; k0 < DM; k0 += 32) {
    v16h a = ld_fragA(Of16 + (size_t)m0 * DM + k0, DM);
#pragma unroll
    for (int j = 0; j < 4; ++j) {
      v16h b = ld_fragB(WT + (size_t)(n0 + 16 * j) * DM + k0, DM);
      acc[j] = wmma_f16(a, b, acc[j]);
    }
  }

  int hi8 = (lane & 16) ? 8 : 0;
  int nl  = lane & 15;
#pragma unroll
  for (int j = 0; j < 4; ++j) {
    float bias = bout[n0 + 16 * j + nl];
#pragma unroll
    for (int r = 0; r < 8; ++r)
      Y[(size_t)(m0 + r + hi8) * DM + n0 + 16 * j + nl] = acc[j][r] + bias;
  }
}

// ---------------- host launcher ----------------
extern "C" void kernel_launch(void* const* d_in, const int* in_sizes, int n_in,
                              void* d_out, int out_size, void* d_ws, size_t ws_size,
                              hipStream_t stream) {
  const float* x    = (const float*)d_in[0];  // [2,2048,1024]
  const float* mask = (const float*)d_in[1];  // [2048,2048]
  const float* Wqkv = (const float*)d_in[2];  // [1024,3072]
  const float* bqkv = (const float*)d_in[3];  // [3072]
  const float* Wout = (const float*)d_in[4];  // [1024,1024]
  const float* bout = (const float*)d_in[5];  // [1024]
  float* out = (float*)d_out;                 // [2,2048,1024]

  char* ws = (char*)d_ws;
  size_t off = 0;
  auto carve = [&](size_t bytes) {
    char* p = ws + off;
    off += (bytes + 255) & ~(size_t)255;
    return p;
  };
  _Float16* xh  = (_Float16*)carve((size_t)4096 * 1024 * 2);
  _Float16* WqT = (_Float16*)carve((size_t)3072 * 1024 * 2);
  _Float16* WoT = (_Float16*)carve((size_t)1024 * 1024 * 2);
  _Float16* Qh  = (_Float16*)carve((size_t)32 * 2048 * 64 * 2);
  _Float16* Kh  = (_Float16*)carve((size_t)32 * 2048 * 64 * 2);
  _Float16* Vth = (_Float16*)carve((size_t)32 * 2048 * 64 * 2);
  _Float16* Oh  = xh;   // x-f16 is dead after qkv_gemm; reuse for attention output

  cvt_f16      <<<16384, 256, 0, stream>>>(x, xh, 4096 * 1024);
  transpose_cvt<<<12288, 256, 0, stream>>>(Wqkv, WqT, 1024, 3072);
  transpose_cvt<<< 4096, 256, 0, stream>>>(Wout, WoT, 1024, 1024);

  qkv_gemm<<<3072, 128, 0, stream>>>(xh, WqT, bqkv, Qh, Kh, Vth);   // 12288 waves
  attn    <<<2048,  64, 0, stream>>>(Qh, Kh, Vth, mask, Oh);        //  4096 waves
  out_gemm<<<1024, 128, 0, stream>>>(Oh, WoT, bout, out);           //  4096 waves
}
